// FAttention1d_43550968382136
// MI455X (gfx1250) — matrix operand order repaired and checked by reading, not compiled, not run on hardware
//
#include <hip/hip_runtime.h>

typedef __attribute__((ext_vector_type(16))) __bf16 v16bf;
typedef __attribute__((ext_vector_type(8)))  float  v8f;

#define S_LEN 2048
#define D_DIM 64
#define L2E   1.4426950408889634f   // log2(e)

__device__ __forceinline__ v8f wmma_bf16(v16bf a, v16bf b, v8f c) {
  // V_WMMA_F32_16X16X32_BF16 : D = A(16x32 bf16) x B(32x16 bf16) + C(16x16 f32)
  return __builtin_amdgcn_wmma_f32_16x16x32_bf16(
      /*neg_a=*/false, a, /*neg_b=*/false, b,
      /*c_mod=*/(short)0, c, /*reuse_a=*/false, /*reuse_b=*/false);
}

// Split 8 consecutive f32 into bf16 hi/lo fragments at elements [E0, E0+8).
template <int E0>
__device__ __forceinline__ void split_run8(const float* p, v16bf& hi, v16bf& lo) {
#pragma unroll
  for (int i = 0; i < 8; ++i) {
    float x = p[i];
    __bf16 h = (__bf16)x;
    hi[E0 + i] = h;
    lo[E0 + i] = (__bf16)(x - (float)h);
  }
}

// 16 contiguous bf16 -> v16bf (B fragment; merges to ds_load_b128 x2)
__device__ __forceinline__ v16bf load16bf(const __bf16* p) {
  v16bf r;
#pragma unroll
  for (int i = 0; i < 16; ++i) r[i] = p[i];
  return r;
}

// A fragment from bf16 LDS: elements 0..7 from p[0..8), 8..15 from p[16..24)
__device__ __forceinline__ v16bf load2run8(const __bf16* p) {
  v16bf r;
#pragma unroll
  for (int i = 0; i < 8; ++i) { r[i] = p[i]; r[8 + i] = p[16 + i]; }
  return r;
}

// ---------------------------------------------------------------------------
// Kernel 1: per-column (k) softmax stats over the QUERY axis, bf16x3 scores.
// Grid (S/128, B*H), 256 thr. Wave w owns columns [kb+16w, +16).
// ---------------------------------------------------------------------------
__global__ __launch_bounds__(256) void colstats_kernel(
    const float* __restrict__ Q, const float* __restrict__ K,
    const int* __restrict__ regp,
    float* __restrict__ m_out, float* __restrict__ s_out)
{
  __shared__ __bf16 ldsQhi[16 * 64];
  __shared__ __bf16 ldsQlo[16 * 64];

  const int bh    = blockIdx.y;
  const int kb    = blockIdx.x * 128;
  const int tid   = threadIdx.x;
  const int wave  = tid >> 5;
  const int lane  = tid & 31;
  const int nl    = lane & 15;
  const int base8 = (lane < 16) ? 0 : 8;    // A-layout K interleave
  const int off16 = (lane < 16) ? 0 : 16;   // contiguous K split, unused in this kernel
  const int hiH   = lane >> 4;
  const float regv = (float)regp[0];

  const float* Qh = Q + (size_t)bh * S_LEN * D_DIM;
  const float* Kh = K + (size_t)bh * S_LEN * D_DIM;

  const int k0 = kb + wave * 16;

  // Resident B fragments (K^T), hi/lo split, 2 chunks of K-dim 32 cover D=64.
  v16bf bKhi[2], bKlo[2];
#pragma unroll
  for (int c = 0; c < 2; ++c) {
    const float* p = Kh + (size_t)(k0 + nl) * D_DIM + 32 * c + base8;
    split_run8<0>(p, bKhi[c], bKlo[c]);
    split_run8<8>(p + 16, bKhi[c], bKlo[c]);
  }

  float m = -__builtin_inff();
  float s = 0.0f;

  const int row  = tid >> 4;
  const int col4 = (tid & 15) * 4;

  for (int qt = 0; qt < S_LEN / 16; ++qt) {
    const int q0 = qt * 16;
    // cooperative load+convert Q tile (16x64) to bf16 hi/lo in LDS
    const float4 qv = *(const float4*)(Qh + (size_t)(q0 + row) * D_DIM + col4);
    const float qa[4] = {qv.x, qv.y, qv.z, qv.w};
#pragma unroll
    for (int i = 0; i < 4; ++i) {
      __bf16 h = (__bf16)qa[i];
      ldsQhi[row * 64 + col4 + i] = h;
      ldsQlo[row * 64 + col4 + i] = (__bf16)(qa[i] - (float)h);
    }
    __syncthreads();

    v8f acc = {};
#pragma unroll
    for (int c = 0; c < 2; ++c) {
      v16bf ahi = load2run8(&ldsQhi[nl * 64 + 32 * c + base8]);
      v16bf alo = load2run8(&ldsQlo[nl * 64 + 32 * c + base8]);
      acc = wmma_bf16(ahi, bKhi[c], acc);   // hi*hi
      acc = wmma_bf16(ahi, bKlo[c], acc);   // hi*lo
      acc = wmma_bf16(alo, bKhi[c], acc);   // lo*hi
    }

    // + reg * eye(S): only possible when this q-tile IS the wave's k block.
    // q0 and k0 are wave-uniform -> scalar branch, taken 1/128 iterations.
    if (q0 == k0) {
#pragma unroll
      for (int j = 0; j < 8; ++j) {
        if (j + 8 * hiH == nl) acc[j] += regv;
      }
    }

    // online column max/sum (column spans this lane's 8 rows + lane^16's 8)
    float mx = acc[0];
#pragma unroll
    for (int j = 1; j < 8; ++j) mx = fmaxf(mx, acc[j]);
    mx = fmaxf(mx, __shfl_xor(mx, 16, 32));

    const float mn  = fmaxf(m, mx);
    const float mnL = mn * L2E;
    float ps = 0.0f;
#pragma unroll
    for (int j = 0; j < 8; ++j)
      ps += __builtin_amdgcn_exp2f(fmaf(acc[j], L2E, -mnL));
    ps += __shfl_xor(ps, 16, 32);
    s = s * __builtin_amdgcn_exp2f(fmaf(m, L2E, -mnL)) + ps;
    m = mn;

    __syncthreads();
  }

  if (lane < 16) {
    m_out[(size_t)bh * S_LEN + k0 + lane] = m;
    s_out[(size_t)bh * S_LEN + k0 + lane] = s;
  }
}

// ---------------------------------------------------------------------------
// Kernel 2: out[q,:] = sum_k exp(att[q,k]-m[k])/s[k] * V[k,:], bf16x3 both
// GEMMs. Grid (S/128, B*H), 256 thr. Wave w owns q rows [qb+16w,+16), k-step 32.
// ---------------------------------------------------------------------------
__global__ __launch_bounds__(256) void attn_out_kernel(
    const float* __restrict__ Q, const float* __restrict__ K,
    const float* __restrict__ V, const int* __restrict__ regp,
    const float* __restrict__ m_in, const float* __restrict__ s_in,
    float* __restrict__ out)
{
  __shared__ __bf16 ldsKhi[32 * 64];   // K tile, row-major (k, d)
  __shared__ __bf16 ldsKlo[32 * 64];
  __shared__ __bf16 ldsVThi[64 * 32];  // V tile, transposed (v, k)
  __shared__ __bf16 ldsVTlo[64 * 32];
  __shared__ float  ldsP[8 * 16 * 32]; // per-wave P tile (f32, C->A staging)
  __shared__ float  ldsM[32];
  __shared__ float  ldsS[32];

  const int bh    = blockIdx.y;
  const int qb    = blockIdx.x * 128;
  const int tid   = threadIdx.x;
  const int wave  = tid >> 5;
  const int lane  = tid & 31;
  const int nl    = lane & 15;
  const int base8 = (lane < 16) ? 0 : 8;
  const int off16 = (lane < 16) ? 0 : 16;
  const int hiH   = lane >> 4;
  const float regv = (float)regp[0];

  const size_t hb = (size_t)bh * S_LEN * D_DIM;
  const float* Qh = Q + hb;
  const float* Kh = K + hb;
  const float* Vh = V + hb;

  const int q0 = qb + wave * 16;

  // Resident A fragments: wave's 16 Q rows x D=64, hi/lo split, 2 chunks.
  v16bf aQhi[2], aQlo[2];
#pragma unroll
  for (int c = 0; c < 2; ++c) {
    const float* p = Qh + (size_t)(q0 + nl) * D_DIM + 32 * c + off16;
    split_run8<0>(p, aQhi[c], aQlo[c]);
    split_run8<8>(p + 8, aQhi[c], aQlo[c]);
  }

  v8f acc[4] = {};   // 16x64 output tile

  float* ldsPw = ldsP + wave * 512;

  for (int kt = 0; kt < S_LEN / 32; ++kt) {
    const int k0 = kt * 32;

    // --- cooperative load + bf16 hi/lo conversion of K (32x64) ---
    {
      const int r  = tid >> 3;          // 0..31
      const int c8 = (tid & 7) * 8;     // 0..56
      const float* p = Kh + (size_t)(k0 + r) * D_DIM + c8;
#pragma unroll
      for (int i = 0; i < 8; ++i) {
        float x = p[i];
        __bf16 h = (__bf16)x;
        ldsKhi[r * 64 + c8 + i] = h;
        ldsKlo[r * 64 + c8 + i] = (__bf16)(x - (float)h);
      }
    }
    // --- V (32x64) transposed to (v=64, k=32), hi/lo bf16 ---
    {
      const int n   = tid & 63;
      const int kb2 = (tid >> 6) * 8;   // 0,8,16,24
#pragma unroll
      for (int i = 0; i < 8; ++i) {
        float x = Vh[(size_t)(k0 + kb2 + i) * D_DIM + n];
        __bf16 h = (__bf16)x;
        ldsVThi[n * 32 + kb2 + i] = h;
        ldsVTlo[n * 32 + kb2 + i] = (__bf16)(x - (float)h);
      }
    }
    if (tid < 32) {
      ldsM[tid] = m_in[(size_t)bh * S_LEN + k0 + tid];
      ldsS[tid] = s_in[(size_t)bh * S_LEN + k0 + tid];
    }
    __syncthreads();

    // --- scores: att[nt] = Q(16x64) x K^T(64x16) for two 16-col tiles ---
    v8f att[2] = {};
#pragma unroll
    for (int nt = 0; nt < 2; ++nt) {
#pragma unroll
      for (int c = 0; c < 2; ++c) {
        v16bf bhi = load16bf(&ldsKhi[(nt * 16 + nl) * 64 + 32 * c + off16]);
        v16bf blo = load16bf(&ldsKlo[(nt * 16 + nl) * 64 + 32 * c + off16]);
        att[nt] = wmma_bf16(aQhi[c], bhi, att[nt]);
        att[nt] = wmma_bf16(aQhi[c], blo, att[nt]);
        att[nt] = wmma_bf16(aQlo[c], bhi, att[nt]);
      }
    }

    // --- P = exp2(x*L2E - (m*L2E + log2(s))) staged f32 in LDS ---
#pragma unroll
    for (int nt = 0; nt < 2; ++nt) {
      const int ncol = nt * 16 + nl;
      // diagonal: wave-uniform branch, taken 1/64 iterations per nt
      if (q0 == k0 + nt * 16) {
#pragma unroll
        for (int j = 0; j < 8; ++j) {
          if (j + 8 * hiH == nl) att[nt][j] += regv;
        }
      }
      const float c2 = fmaf(ldsM[ncol], L2E, __builtin_amdgcn_logf(ldsS[ncol]));
#pragma unroll
      for (int j = 0; j < 8; ++j) {
        const int M = j + 8 * hiH;
        ldsPw[M * 32 + ncol] =
            __builtin_amdgcn_exp2f(fmaf(att[nt][j], L2E, -c2));
      }
    }
    __syncthreads();

    // --- out += P(16x32) x V(32x64), bf16x3, one K=32 chunk, 4 N tiles ---
    v16bf aPhi, aPlo;
    split_run8<0>(&ldsPw[nl * 32 + off16], aPhi, aPlo);
    split_run8<8>(&ldsPw[nl * 32 + off16 + 8], aPhi, aPlo);
#pragma unroll
    for (int nt = 0; nt < 4; ++nt) {
      v16bf bhi = load16bf(&ldsVThi[(nt * 16 + nl) * 32 + off16]);
      v16bf blo = load16bf(&ldsVTlo[(nt * 16 + nl) * 32 + off16]);
      acc[nt] = wmma_bf16(aPhi, bhi, acc[nt]);
      acc[nt] = wmma_bf16(aPhi, blo, acc[nt]);
      acc[nt] = wmma_bf16(aPlo, bhi, acc[nt]);
    }
    __syncthreads();
  }

  float* outh = out + hb;
#pragma unroll
  for (int nt = 0; nt < 4; ++nt) {
#pragma unroll
    for (int j = 0; j < 8; ++j) {
      const int M = j + 8 * hiH;
      outh[(size_t)(q0 + M) * D_DIM + nt * 16 + nl] = acc[nt][j];
    }
  }
}

extern "C" void kernel_launch(void* const* d_in, const int* in_sizes, int n_in,
                              void* d_out, int out_size, void* d_ws, size_t ws_size,
                              hipStream_t stream) {
  const float* Q   = (const float*)d_in[0];
  const float* K   = (const float*)d_in[1];
  const float* V   = (const float*)d_in[2];
  const int*   reg = (const int*)d_in[3];
  float* out = (float*)d_out;

  const int BH = in_sizes[0] / (S_LEN * D_DIM);   // B*H = 32

  float* m_ws = (float*)d_ws;
  float* s_ws = m_ws + (size_t)BH * S_LEN;

  dim3 grid(S_LEN / 128, BH);
  colstats_kernel<<<grid, 256, 0, stream>>>(Q, K, reg, m_ws, s_ws);
  attn_out_kernel<<<grid, 256, 0, stream>>>(Q, K, V, reg, m_ws, s_ws, out);
}
